// PointNet2Encoder_47682726920300
// MI455X (gfx1250) — matrix operand order repaired and checked by reading, not compiled, not run on hardware
//
#include <hip/hip_runtime.h>

typedef __attribute__((ext_vector_type(16))) _Float16 v16h;
typedef __attribute__((ext_vector_type(8)))  _Float16 v8h;
typedef __attribute__((ext_vector_type(8)))  float    v8f;
typedef __attribute__((ext_vector_type(4)))  float    v4f;

#define DEV __device__ __forceinline__

// ---------------------------------------------------------------------------
// WMMA fragment helpers (layouts per CDNA5 ISA §7.12.2, wave32)
// ---------------------------------------------------------------------------

// A-matrix (16x32 f16): lane<16 holds row=lane, K={0..7,16..23}; lane>=16 holds
// row=lane-16, K={8..15,24..31}.  Element e -> K offset within the 32-K tile:
DEV int a_k(int e, int lane) { return ((e < 8) ? e : e + 8) + ((lane & 16) ? 8 : 0); }

DEV v16h cat8(v8h lo, v8h hi) {
  return __builtin_shufflevector(lo, hi, 0, 1, 2, 3, 4, 5, 6, 7, 8, 9, 10, 11, 12, 13, 14, 15);
}

DEV v8f wmma_f16(v16h a, v16h b, v8f c) {
  return __builtin_amdgcn_wmma_f32_16x16x32_f16(false, a, false, b, (short)0, c, false, false);
}

// B-matrix fragments are pre-swizzled by the prep kernels: fragment f holds
// 32 lanes x 16 contiguous halves -> one 32-byte load per lane (2x b128).
DEV v16h load_bfrag(const _Float16* __restrict__ swz, int frag, int lane) {
  const _Float16* p = swz + ((size_t)frag * 32 + lane) * 16;
  return cat8(*(const v8h*)p, *(const v8h*)(p + 8));
}

// A fragment from a row-major LDS tile [16][pitch] (f16, 16B-aligned rows).
// Element groups e=0..7 / e=8..15 are contiguous K runs -> 2x ds_load_b128.
DEV v16h load_afrag_lds(const _Float16* act, int pitch, int kt, int lane) {
  const _Float16* p = act + (lane & 15) * pitch + kt * 32 + ((lane & 16) ? 8 : 0);
  return cat8(*(const v8h*)p, *(const v8h*)(p + 16));
}

// C/D (16x16 f32): lane holds col (lane&15); VGPR v is row v + 8*(lane>=16).
// Store bias+ReLU result into a row-major f16 LDS tile for the next layer.
DEV void store_act(v8f d, const float* __restrict__ bias, int nt,
                   _Float16* act, int pitch, int lane) {
  const int col = nt * 16 + (lane & 15);
  const float bb = bias[col];
  const int r0 = (lane & 16) ? 8 : 0;
#pragma unroll
  for (int v = 0; v < 8; ++v) {
    float x = d[v] + bb;
    x = x > 0.f ? x : 0.f;
    act[(r0 + v) * pitch + col] = (_Float16)x;
  }
}

// ---------------------------------------------------------------------------
// Farthest point sampling: one workgroup per batch, xyz staged in LDS
// (SA1: 192KB of LDS -- uses CDNA5's 320KB/WGP).  1024 threads, wave32
// shuffle argmax + cross-wave LDS reduction.
// ---------------------------------------------------------------------------
template <int N, int NPOINT, int XSTR>
__global__ __launch_bounds__(1024) void fps_kernel(const float* __restrict__ xyz,
                                                   float* __restrict__ new_xyz) {
  extern __shared__ float sm[];
  float* X = sm;
  float* Y = sm + N;
  float* Z = sm + 2 * N;
  __shared__ float swv[32];
  __shared__ int   swi[32];
  __shared__ int   sfar;
  __shared__ int   sfi[NPOINT];

  const int b = blockIdx.x;
  const int tid = threadIdx.x;
  const float* src = xyz + (size_t)b * N * XSTR;
  for (int i = tid; i < N; i += 1024) {
    const float* p = src + (size_t)i * XSTR;
    X[i] = p[0]; Y[i] = p[1]; Z[i] = p[2];
  }
  constexpr int PPT = N / 1024;
  float dists[PPT];
#pragma unroll
  for (int j = 0; j < PPT; ++j) dists[j] = 1e10f;
  __syncthreads();

  const int lane = tid & 31, wv = tid >> 5;
  int far = 0;
  for (int it = 0; it < NPOINT; ++it) {
    if (tid == 0) sfi[it] = far;
    const float fx = X[far], fy = Y[far], fz = Z[far];
    float best = -1.f; int bi = 0;
#pragma unroll
    for (int j = 0; j < PPT; ++j) {
      const int i = tid + j * 1024;
      const float dx = X[i] - fx, dy = Y[i] - fy, dz = Z[i] - fz;
      const float d = dx * dx + dy * dy + dz * dz;
      const float dd = dists[j] = fminf(dists[j], d);
      if (dd > best) { best = dd; bi = i; }
    }
#pragma unroll
    for (int off = 16; off > 0; off >>= 1) {
      const float ov = __shfl_down(best, off, 32);
      const int   oi = __shfl_down(bi, off, 32);
      if (ov > best) { best = ov; bi = oi; }
    }
    if (lane == 0) { swv[wv] = best; swi[wv] = bi; }
    __syncthreads();
    if (wv == 0) {
      best = swv[lane]; bi = swi[lane];
#pragma unroll
      for (int off = 16; off > 0; off >>= 1) {
        const float ov = __shfl_down(best, off, 32);
        const int   oi = __shfl_down(bi, off, 32);
        if (ov > best) { best = ov; bi = oi; }
      }
      if (lane == 0) sfar = bi;
    }
    __syncthreads();
    far = sfar;
  }
  for (int it = tid; it < NPOINT; it += 1024) {
    const int id = sfi[it];
    float* o = new_xyz + ((size_t)b * NPOINT + it) * 3;
    o[0] = X[id]; o[1] = Y[id]; o[2] = Z[id];
  }
}

// ---------------------------------------------------------------------------
// Dual-radius ball query: one wave per query; ordered first-K collection via
// ballot/popc (matches reference's smallest-index top_k + first-hit padding).
// ---------------------------------------------------------------------------
template <int NS1, int NS2>
__global__ __launch_bounds__(256) void ball_query_kernel(
    const float* __restrict__ xyz, int xstride, const float* __restrict__ new_xyz,
    int* __restrict__ g1, int* __restrict__ g2,
    float r1sq, float r2sq, int npoint, int N) {
  const int lane = threadIdx.x & 31;
  const int wave = threadIdx.x >> 5;
  const int q = blockIdx.x * 8 + wave;
  const int b = q / npoint;
  const float* nx = new_xyz + (size_t)q * 3;
  const float cx = nx[0], cy = nx[1], cz = nx[2];
  const float* src = xyz + (size_t)b * N * xstride;
  int* o1 = g1 + (size_t)q * NS1;
  int* o2 = g2 + (size_t)q * NS2;
  int c1 = 0, c2 = 0, f1i = 0, f2i = 0;
  for (int base = 0; base < N; base += 32) {
    const int i = base + lane;
    const float* p = src + (size_t)i * xstride;
    const float dx = p[0] - cx, dy = p[1] - cy, dz = p[2] - cz;
    const float d = dx * dx + dy * dy + dz * dz;
    const unsigned m2 = (unsigned)__ballot(d < r2sq);
    const unsigned m1 = (unsigned)__ballot(d < r1sq);
    const unsigned below = (1u << lane) - 1u;
    if (c1 < NS1 && m1) {
      if (c1 == 0) f1i = base + (__ffs((int)m1) - 1);
      if (d < r1sq) {
        const int pos = c1 + __popc(m1 & below);
        if (pos < NS1) o1[pos] = i;
      }
      c1 += __popc(m1);
    }
    if (c2 < NS2 && m2) {
      if (c2 == 0) f2i = base + (__ffs((int)m2) - 1);
      if (d < r2sq) {
        const int pos = c2 + __popc(m2 & below);
        if (pos < NS2) o2[pos] = i;
      }
      c2 += __popc(m2);
    }
    if (c1 >= NS1 && c2 >= NS2) break;
  }
  c1 = c1 < NS1 ? c1 : NS1;
  c2 = c2 < NS2 ? c2 : NS2;
  for (int p = c1 + lane; p < NS1; p += 32) o1[p] = f1i;
  for (int p = c2 + lane; p < NS2; p += 32) o2[p] = f2i;
}

// ---------------------------------------------------------------------------
// Weight prep: fold BN into weights, emit f16 in *B-fragment-swizzled* order:
// Wswz[((kt*NT+nt)*32 + lane)*16 + e] = W'[kt*32 + 8*(lane>=16) + e + 8*(e>=8)][nt*16+(lane&15)]
// Layer-1 weights get their input rows permuted to [feats.., centered-xyz]
// so the gather kernel can use aligned vector loads on the feature block.
// ---------------------------------------------------------------------------
__global__ void prep_bn_weights(const float* __restrict__ W,
                                const float* __restrict__ gamma, const float* __restrict__ beta,
                                const float* __restrict__ mean, const float* __restrict__ var,
                                int cin, int cout, int KT, int NT, int perm,
                                _Float16* __restrict__ Wswz, float* __restrict__ bias) {
  const int i = blockIdx.x * blockDim.x + threadIdx.x;
  if (i < cout) {
    const float s = gamma[i] * rsqrtf(var[i] + 1e-5f);
    bias[i] = beta[i] - mean[i] * s;
  }
  if (i < KT * NT * 512) {
    const int f = i >> 9;
    const int l = (i >> 4) & 31;
    const int e = i & 15;
    const int kt = f / NT, nt = f % NT;
    const int n = nt * 16 + (l & 15);
    const int k = kt * 32 + ((l & 16) ? 8 : 0) + ((e < 8) ? e : e + 8);
    float w = 0.f;
    if (k < cin) {
      int ks = k;
      if (perm) ks = (k < cin - 3) ? (k + 3) : (k - (cin - 3));
      const float s = gamma[n] * rsqrtf(var[n] + 1e-5f);
      w = W[(size_t)ks * cout + n] * s;
    }
    Wswz[i] = (_Float16)w;
  }
}

__global__ void prep_lin_weights(const float* __restrict__ W, const float* __restrict__ bvec,
                                 _Float16* __restrict__ Wswz, float* __restrict__ bias) {
  const int i = blockIdx.x * blockDim.x + threadIdx.x;
  if (i < 128) bias[i] = (i < 125) ? bvec[i] : 0.f;
  if (i < 8 * 8 * 512) {
    const int f = i >> 9;
    const int l = (i >> 4) & 31;
    const int e = i & 15;
    const int kt = f / 8, nt = f % 8;
    const int n = nt * 16 + (l & 15);
    const int k = kt * 32 + ((l & 16) ? 8 : 0) + ((e < 8) ? e : e + 8);
    Wswz[i] = (_Float16)((n < 125) ? W[(size_t)k * 125 + n] : 0.f);
  }
}

// ---------------------------------------------------------------------------
// Grouped 3-layer shared-MLP + maxpool, fully WMMA.  One wave per query point;
// NS samples = NS/16 M-tiles.  Inter-layer relayout via per-wave LDS tiles
// (DS in-order per wave -> no barriers).  Input channels are ordered
// [feats(CIN-3), centered-xyz(3)] to match the permuted layer-1 weights.
// ---------------------------------------------------------------------------
template <int NS, int CIN, int C1, int C2, int C3>
__global__ __launch_bounds__(256) void group_mlp_kernel(
    const int* __restrict__ gidx, const float* __restrict__ xyz, int xstride,
    const float* __restrict__ feat, int fstride, const float* __restrict__ new_xyz,
    const _Float16* __restrict__ W1, const float* __restrict__ B1,
    const _Float16* __restrict__ W2, const float* __restrict__ B2,
    const _Float16* __restrict__ W3, const float* __restrict__ B3,
    float* __restrict__ fout, int fostride, int coloff, int npoint, int N) {
  constexpr int NF = CIN - 3;
  constexpr int KT1 = (CIN + 31) / 32;
  constexpr int NT1 = C1 / 16, KT2 = (C1 + 31) / 32;
  constexpr int NT2 = C2 / 16, KT3 = (C2 + 31) / 32;
  constexpr int NT3 = C3 / 16;
  constexpr int PITCH = (KT2 > KT3 ? KT2 : KT3) * 32;
  constexpr int MT = NS / 16;
  __shared__ __align__(32) _Float16 smem[8 * 2 * 16 * PITCH];

  const int lane = threadIdx.x & 31;
  const int wave = threadIdx.x >> 5;
  const int q = blockIdx.x * 8 + wave;
  const int b = q / npoint;
  _Float16* act0 = smem + (size_t)wave * 2 * 16 * PITCH;
  _Float16* act1 = act0 + 16 * PITCH;
  for (int i = lane; i < 2 * 16 * PITCH; i += 32) act0[i] = (_Float16)0.f;  // zero K-pad

  const int* grow = gidx + (size_t)q * NS;
  const float* nx = new_xyz + (size_t)q * 3;
  float ctr[3];
  ctr[0] = nx[0]; ctr[1] = nx[1]; ctr[2] = nx[2];
  const float* xyz_b  = xyz + (size_t)b * N * xstride;
  const float* feat_b = feat + (size_t)b * N * fstride;

  float vmax[NT3];
#pragma unroll
  for (int n = 0; n < NT3; ++n) vmax[n] = 0.f;  // ReLU outputs are >= 0

  for (int mt = 0; mt < MT; ++mt) {
    // layer 1: gather [feats ++ centered-xyz] directly into A fragments
    {
      const int row = grow[mt * 16 + (lane & 15)];
      const float* xr = xyz_b + (size_t)row * xstride;
      const float* fr = feat_b + (size_t)row * fstride;
      v16h a[KT1];
#pragma unroll
      for (int kt = 0; kt < KT1; ++kt) {
        if ((kt + 1) * 32 <= NF) {
          // pure-feature tile, rows are 16B-aligned -> vector loads
          const float* pr = fr + kt * 32 + ((lane & 16) ? 8 : 0);
          const v4f x0 = *(const v4f*)pr;
          const v4f x1 = *(const v4f*)(pr + 4);
          const v4f x2 = *(const v4f*)(pr + 16);
          const v4f x3 = *(const v4f*)(pr + 20);
#pragma unroll
          for (int j = 0; j < 4; ++j) {
            a[kt][j]      = (_Float16)x0[j];
            a[kt][4 + j]  = (_Float16)x1[j];
            a[kt][8 + j]  = (_Float16)x2[j];
            a[kt][12 + j] = (_Float16)x3[j];
          }
        } else {
#pragma unroll
          for (int e = 0; e < 16; ++e) {
            const int c = kt * 32 + a_k(e, lane);
            float v = 0.f;
            if (c < CIN) v = (c < NF) ? fr[c] : (xr[c - NF] - ctr[c - NF]);
            a[kt][e] = (_Float16)v;
          }
        }
      }
#pragma unroll
      for (int nt = 0; nt < NT1; ++nt) {
        v8f acc = {};
#pragma unroll
        for (int kt = 0; kt < KT1; ++kt)
          acc = wmma_f16(a[kt], load_bfrag(W1, kt * NT1 + nt, lane), acc);
        store_act(acc, B1, nt, act0, PITCH, lane);
      }
    }
    // layer 2: act0 -> act1
    {
      v16h a[KT2];
#pragma unroll
      for (int kt = 0; kt < KT2; ++kt) a[kt] = load_afrag_lds(act0, PITCH, kt, lane);
#pragma unroll
      for (int nt = 0; nt < NT2; ++nt) {
        v8f acc = {};
#pragma unroll
        for (int kt = 0; kt < KT2; ++kt)
          acc = wmma_f16(a[kt], load_bfrag(W2, kt * NT2 + nt, lane), acc);
        store_act(acc, B2, nt, act1, PITCH, lane);
      }
    }
    // layer 3: act1 -> running max
    {
      v16h a[KT3];
#pragma unroll
      for (int kt = 0; kt < KT3; ++kt) a[kt] = load_afrag_lds(act1, PITCH, kt, lane);
#pragma unroll
      for (int nt = 0; nt < NT3; ++nt) {
        v8f acc = {};
#pragma unroll
        for (int kt = 0; kt < KT3; ++kt)
          acc = wmma_f16(a[kt], load_bfrag(W3, kt * NT3 + nt, lane), acc);
        const float bb = B3[nt * 16 + (lane & 15)];
        float m = 0.f;
#pragma unroll
        for (int v = 0; v < 8; ++v) {
          float x = acc[v] + bb;
          x = x > 0.f ? x : 0.f;
          m = fmaxf(m, x);
        }
        m = fmaxf(m, __shfl_xor(m, 16, 32));
        vmax[nt] = fmaxf(vmax[nt], m);
      }
    }
  }
  if (lane < 16) {
    float* o = fout + (size_t)q * fostride + coloff;
#pragma unroll
    for (int nt = 0; nt < NT3; ++nt) o[nt * 16 + lane] = vmax[nt];
  }
}

// ---------------------------------------------------------------------------
// Final projection (2048x256 @ 256x128pad) + xyz concat, WMMA.
// ---------------------------------------------------------------------------
__global__ __launch_bounds__(256) void final_proj_kernel(
    const float* __restrict__ f2, const _Float16* __restrict__ Wswz,
    const float* __restrict__ bias, const float* __restrict__ nxyz2,
    float* __restrict__ out) {
  const int lane = threadIdx.x & 31;
  const int wave = threadIdx.x >> 5;
  const int rb = (blockIdx.x * 8 + wave) * 16;
  v16h a[8];
  const float* fr = f2 + (size_t)(rb + (lane & 15)) * 256;
#pragma unroll
  for (int kt = 0; kt < 8; ++kt) {
    const float* pr = fr + kt * 32 + ((lane & 16) ? 8 : 0);
    const v4f x0 = *(const v4f*)pr;
    const v4f x1 = *(const v4f*)(pr + 4);
    const v4f x2 = *(const v4f*)(pr + 16);
    const v4f x3 = *(const v4f*)(pr + 20);
#pragma unroll
    for (int j = 0; j < 4; ++j) {
      a[kt][j]      = (_Float16)x0[j];
      a[kt][4 + j]  = (_Float16)x1[j];
      a[kt][8 + j]  = (_Float16)x2[j];
      a[kt][12 + j] = (_Float16)x3[j];
    }
  }
#pragma unroll
  for (int nt = 0; nt < 8; ++nt) {
    v8f acc = {};
#pragma unroll
    for (int kt = 0; kt < 8; ++kt)
      acc = wmma_f16(a[kt], load_bfrag(Wswz, kt * 8 + nt, lane), acc);
    const int col = nt * 16 + (lane & 15);
    if (col < 125) {
      const float bb = bias[col];
      const int r0 = rb + ((lane & 16) ? 8 : 0);
#pragma unroll
      for (int v = 0; v < 8; ++v)
        out[(size_t)(r0 + v) * 128 + 3 + col] = acc[v] + bb;
    }
  }
  if (lane < 16) {
    const int row = rb + lane;
    out[(size_t)row * 128 + 0] = nxyz2[(size_t)row * 3 + 0];
    out[(size_t)row * 128 + 1] = nxyz2[(size_t)row * 3 + 1];
    out[(size_t)row * 128 + 2] = nxyz2[(size_t)row * 3 + 2];
  }
}

// ---------------------------------------------------------------------------
// Orchestration
// ---------------------------------------------------------------------------
extern "C" void kernel_launch(void* const* d_in, const int* in_sizes, int n_in,
                              void* d_out, int out_size, void* d_ws, size_t ws_size,
                              hipStream_t stream) {
  (void)in_sizes; (void)n_in; (void)out_size; (void)ws_size;
  // Input flattening (setup_inputs insertion order, depth-first):
  // [0]=pointcloud; then 12 layers x {W,gamma,beta,mean,var}; [61]=lin_w; [62]=lin_b
  const float* PC = (const float*)d_in[0];
  const float *LW[12], *LG[12], *LB[12], *LM[12], *LV[12];
  for (int l = 0; l < 12; ++l) {
    LW[l] = (const float*)d_in[1 + l * 5 + 0];
    LG[l] = (const float*)d_in[1 + l * 5 + 1];
    LB[l] = (const float*)d_in[1 + l * 5 + 2];
    LM[l] = (const float*)d_in[1 + l * 5 + 3];
    LV[l] = (const float*)d_in[1 + l * 5 + 4];
  }
  const float* LINW = (const float*)d_in[61];
  const float* LINB = (const float*)d_in[62];

  static const int LD[12][5] = {  // {cin, cout, kpad, perm}
      {9, 16, 32, 1},   {16, 16, 32, 0},  {16, 32, 32, 0},
      {9, 32, 32, 1},   {32, 32, 32, 0},  {32, 64, 32, 0},
      {99, 64, 128, 1}, {64, 64, 64, 0},  {64, 128, 64, 0},
      {99, 64, 128, 1}, {64, 96, 64, 0},  {96, 128, 96, 0}};

  char* p = (char*)d_ws;
  auto alloc = [&](size_t bytes) -> void* {
    void* r = (void*)p;
    p += (bytes + 255) & ~(size_t)255;
    return r;
  };

  _Float16* WH[12]; float* BS[12];
  for (int l = 0; l < 12; ++l) {
    WH[l] = (_Float16*)alloc((size_t)LD[l][2] * LD[l][1] * sizeof(_Float16));
    BS[l] = (float*)alloc((size_t)LD[l][1] * sizeof(float));
  }
  _Float16* LINWH = (_Float16*)alloc(256 * 128 * sizeof(_Float16));
  float* LINBS = (float*)alloc(128 * sizeof(float));
  float* NX1 = (float*)alloc((size_t)8 * 1024 * 3 * sizeof(float));
  int*   G1A = (int*)alloc((size_t)8 * 1024 * 16 * sizeof(int));
  int*   G1B = (int*)alloc((size_t)8 * 1024 * 32 * sizeof(int));
  float* F1  = (float*)alloc((size_t)8 * 1024 * 96 * sizeof(float));
  float* NX2 = (float*)alloc((size_t)8 * 256 * 3 * sizeof(float));
  int*   G2A = (int*)alloc((size_t)8 * 256 * 16 * sizeof(int));
  int*   G2B = (int*)alloc((size_t)8 * 256 * 32 * sizeof(int));
  float* F2  = (float*)alloc((size_t)8 * 256 * 256 * sizeof(float));

  for (int l = 0; l < 12; ++l) {
    const int KT = LD[l][2] / 32, NT = LD[l][1] / 16;
    const int total = KT * NT * 512;
    prep_bn_weights<<<(total + 255) / 256, 256, 0, stream>>>(
        LW[l], LG[l], LB[l], LM[l], LV[l], LD[l][0], LD[l][1], KT, NT, LD[l][3],
        WH[l], BS[l]);
  }
  prep_lin_weights<<<(8 * 8 * 512 + 255) / 256, 256, 0, stream>>>(LINW, LINB, LINWH, LINBS);

  // ---- SA1 ----
  fps_kernel<16384, 1024, 9><<<8, 1024, 3 * 16384 * sizeof(float), stream>>>(PC, NX1);
  ball_query_kernel<16, 32><<<(8 * 1024) / 8, 256, 0, stream>>>(
      PC, 9, NX1, G1A, G1B, 0.05f * 0.05f, 0.1f * 0.1f, 1024, 16384);
  group_mlp_kernel<16, 9, 16, 16, 32><<<(8 * 1024) / 8, 256, 0, stream>>>(
      G1A, PC, 9, PC + 3, 9, NX1, WH[0], BS[0], WH[1], BS[1], WH[2], BS[2],
      F1, 96, 0, 1024, 16384);
  group_mlp_kernel<32, 9, 32, 32, 64><<<(8 * 1024) / 8, 256, 0, stream>>>(
      G1B, PC, 9, PC + 3, 9, NX1, WH[3], BS[3], WH[4], BS[4], WH[5], BS[5],
      F1, 96, 32, 1024, 16384);

  // ---- SA2 ----
  fps_kernel<1024, 256, 3><<<8, 1024, 3 * 1024 * sizeof(float), stream>>>(NX1, NX2);
  ball_query_kernel<16, 32><<<(8 * 256) / 8, 256, 0, stream>>>(
      NX1, 3, NX2, G2A, G2B, 0.1f * 0.1f, 0.2f * 0.2f, 256, 1024);
  group_mlp_kernel<16, 99, 64, 64, 128><<<(8 * 256) / 8, 256, 0, stream>>>(
      G2A, NX1, 3, F1, 96, NX2, WH[6], BS[6], WH[7], BS[7], WH[8], BS[8],
      F2, 256, 0, 256, 1024);
  group_mlp_kernel<32, 99, 64, 96, 128><<<(8 * 256) / 8, 256, 0, stream>>>(
      G2B, NX1, 3, F1, 96, NX2, WH[9], BS[9], WH[10], BS[10], WH[11], BS[11],
      F2, 256, 128, 256, 1024);

  // ---- final projection + xyz concat ----
  final_proj_kernel<<<16, 256, 0, stream>>>(F2, LINWH, LINBS, NX2, (float*)d_out);
}
